// BitFeedForward_39548058862238
// MI455X (gfx1250) — 8-bit matrix operand order repaired and checked by reading, compiled, not run on hardware
//
#include <hip/hip_runtime.h>
#include <cstdint>
#include <cstddef>


typedef int v8i __attribute__((ext_vector_type(8)));

#define EPSQ 1e-5f


__global__ void zero_sums_kernel(float* sums) {
    if (threadIdx.x < 2) sums[threadIdx.x] = 0.0f;
}

__global__ __launch_bounds__(256)
void abs_sum_kernel(const float* __restrict__ w, float* __restrict__ sum, size_t n) {
    __shared__ float red[256];
    float s = 0.0f;
    for (size_t i = (size_t)blockIdx.x * blockDim.x + threadIdx.x; i < n;
         i += (size_t)gridDim.x * blockDim.x)
        s += fabsf(w[i]);
    red[threadIdx.x] = s;
    __syncthreads();
    for (int t = 128; t > 0; t >>= 1) {
        if ((int)threadIdx.x < t) red[threadIdx.x] += red[threadIdx.x + t];
        __syncthreads();
    }
    if (threadIdx.x == 0) atomicAdd(sum, red[0]);
}


__global__ __launch_bounds__(256)
void weight_quant_kernel(const float* __restrict__ w, int8_t* __restrict__ wq,
                         const float* __restrict__ sumptr, float inv_count, size_t n) {
    const float mean  = fmaxf(*sumptr * inv_count, EPSQ);
    const float scale = 1.0f / mean;
    for (size_t i = (size_t)blockIdx.x * blockDim.x + threadIdx.x; i < n;
         i += (size_t)gridDim.x * blockDim.x) {
        float q = rintf(w[i] * scale);
        q = fminf(fmaxf(q, -1.0f), 1.0f);
        wq[i] = (int8_t)q;
    }
}


__global__ __launch_bounds__(256)
void act_quant_kernel(const float* __restrict__ src, int8_t* __restrict__ dst,
                      float* __restrict__ scales, int cols) {
    const int row = blockIdx.x;
    const float* r = src + (size_t)row * cols;
    __shared__ float red[256];
    float m = 0.0f;
    for (int c = threadIdx.x; c < cols; c += blockDim.x) m = fmaxf(m, fabsf(r[c]));
    red[threadIdx.x] = m;
    __syncthreads();
    for (int t = 128; t > 0; t >>= 1) {
        if ((int)threadIdx.x < t) red[threadIdx.x] = fmaxf(red[threadIdx.x], red[threadIdx.x + t]);
        __syncthreads();
    }
    const float scale = 127.0f / fmaxf(red[0], EPSQ);
    if (threadIdx.x == 0) scales[row] = scale;
    int8_t* d = dst + (size_t)row * cols;
    for (int c = threadIdx.x; c < cols; c += blockDim.x) {
        float q = rintf(r[c] * scale);
        q = fminf(fmaxf(q, -128.0f), 127.0f);
        d[c] = (int8_t)q;
    }
}


__global__ __launch_bounds__(256)
void gemm_q8_wmma_kernel(const int8_t* __restrict__ A, const int8_t* __restrict__ B,
                         float* __restrict__ out, const float* __restrict__ sx,
                         const float* __restrict__ wsum, float inv_count,
                         int M, int N, int K, int apply_gelu) {
    const int lane = threadIdx.x & 31;
    const int wave = threadIdx.x >> 5;
    const int tilesN = N >> 6;
    const int gw    = blockIdx.x * 8 + wave;
    const int tileM = gw / tilesN;
    const int tileN = gw % tilesN;
    const int m0 = tileM << 4;
    const int n0 = tileN << 6;
    if (m0 >= M) return;

    const int row  = lane & 15;
    const int half = lane >> 4;

    v8i acc0 = {}, acc1 = {}, acc2 = {}, acc3 = {};

    const int8_t* aBase = A + (size_t)(m0 + row) * K + half * 16;
    const int8_t* bBase = B + (size_t)(n0 + row) * K + half * 16;
    const size_t  bStep = (size_t)16 * K;

    for (int k0 = 0; k0 < K; k0 += 64) {
        v8i a;
        {
            const int8_t* p = aBase + k0;
            int4 q0 = *(const int4*)(p);
            int4 q1 = *(const int4*)(p + 32);
            a[0] = q0.x; a[1] = q0.y; a[2] = q0.z; a[3] = q0.w;
            a[4] = q1.x; a[5] = q1.y; a[6] = q1.z; a[7] = q1.w;
        }
        __builtin_prefetch(aBase + k0 + 64, 0, 1);
        __builtin_prefetch(bBase + k0 + 64, 0, 1);

        #define LOAD_B(nb, dst)                                                 \
            {                                                                   \
                const int8_t* p = bBase + (size_t)(nb) * bStep + k0;            \
                int4 q0 = *(const int4*)(p);                                    \
                int4 q1 = *(const int4*)(p + 32);                               \
                dst[0] = q0.x; dst[1] = q0.y; dst[2] = q0.z; dst[3] = q0.w;     \
                dst[4] = q1.x; dst[5] = q1.y; dst[6] = q1.z; dst[7] = q1.w;     \
            }
        v8i b0, b1, b2, b3;
        LOAD_B(0, b0); LOAD_B(1, b1); LOAD_B(2, b2); LOAD_B(3, b3);
        #undef LOAD_B

        acc0 = __builtin_amdgcn_wmma_i32_16x16x64_iu8(true, a, true, b0, acc0, false, false);
        acc1 = __builtin_amdgcn_wmma_i32_16x16x64_iu8(true, a, true, b1, acc1, false, false);
        acc2 = __builtin_amdgcn_wmma_i32_16x16x64_iu8(true, a, true, b2, acc2, false, false);
        acc3 = __builtin_amdgcn_wmma_i32_16x16x64_iu8(true, a, true, b3, acc3, false, false);
    }

    const float wmean = fmaxf(*wsum * inv_count, EPSQ);
    const int   nn    = lane & 15;
    v8i accs[4] = {acc0, acc1, acc2, acc3};

    float rowFactor[8];
    #pragma unroll
    for (int r = 0; r < 8; ++r)
        rowFactor[r] = wmean / sx[m0 + r + half * 8];

    #pragma unroll
    for (int nb = 0; nb < 4; ++nb) {
        const int n = n0 + nb * 16 + nn;
        #pragma unroll
        for (int r = 0; r < 8; ++r) {
            const int m = m0 + r + half * 8;
            float val = (float)accs[nb][r] * rowFactor[r];
            if (apply_gelu)
                val = 0.5f * val * (1.0f + erff(val * 0.7071067811865476f));
            out[(size_t)m * N + n] = val;
        }
    }
}


extern "C" void kernel_launch(void* const* d_in, const int* in_sizes, int n_in,
                              void* d_out, int out_size, void* d_ws, size_t ws_size,
                              hipStream_t stream) {
    const float* x  = (const float*)d_in[0];
    const float* w1 = (const float*)d_in[1];
    const float* w2 = (const float*)d_in[2];
    float* out = (float*)d_out;

    const int D = 2048, H = 8192;
    const int Mrows = 4 * 2048;
    const size_t WN = (size_t)H * D;

    char* ws = (char*)d_ws;
    float*  sums = (float*)ws;
    float*  sx   = (float*)(ws + 4096);
    float*  sh   = (float*)(ws + 36864);
    int8_t* xq   = (int8_t*)(ws + 131072);
    int8_t* w1q  = xq  + (size_t)Mrows * D;
    int8_t* w2q  = w1q + WN;
    int8_t* hq   = w2q + WN;
    float*  h    = (float*)(hq + (size_t)Mrows * H);

    const float invWN = 1.0f / (float)WN;

    zero_sums_kernel<<<1, 32, 0, stream>>>(sums);
    abs_sum_kernel<<<4096, 256, 0, stream>>>(w1, sums + 0, WN);
    abs_sum_kernel<<<4096, 256, 0, stream>>>(w2, sums + 1, WN);

    weight_quant_kernel<<<4096, 256, 0, stream>>>(w1, w1q, sums + 0, invWN, WN);
    weight_quant_kernel<<<4096, 256, 0, stream>>>(w2, w2q, sums + 1, invWN, WN);

    act_quant_kernel<<<Mrows, 256, 0, stream>>>(x, xq, sx, D);

    {
        const int waves = (Mrows / 16) * (H / 64);
        gemm_q8_wmma_kernel<<<waves / 8, 256, 0, stream>>>(
            xq, w1q, h, sx, sums + 0, invWN, Mrows, H, D,  1);
    }

    act_quant_kernel<<<Mrows, 256, 0, stream>>>(h, hq, sh, H);

    {
        const int waves = (Mrows / 16) * (D / 64);
        gemm_q8_wmma_kernel<<<waves / 8, 256, 0, stream>>>(
            hq, w2q, out, sh, sums + 1, invWN, Mrows, D, H,  0);
    }
}
